// Attention_12300786336061
// MI455X (gfx1250) — compile-verified, matrix operand order checked by reading, not run on hardware
//
#include <hip/hip_runtime.h>

#define B_ 128
#define S_ 1024
#define D_ 512

#define SROWS 32            // s-rows per block in k_scores
#define LDP   520           // padded LDS row pitch in bf16 (16B-aligned, bank-staggered)

typedef __attribute__((ext_vector_type(16))) __bf16 v16bf;
typedef __attribute__((ext_vector_type(8)))  __bf16 v8bf;
typedef __attribute__((ext_vector_type(8)))  float  v8f;

__device__ __forceinline__ __bf16 f2bf(float f) {
  union { float f; unsigned u; } v; v.f = f;
  unsigned r = v.u + 0x7FFFu + ((v.u >> 16) & 1u);   // round-to-nearest-even
  union { unsigned short s; __bf16 b; } o; o.s = (unsigned short)(r >> 16);
  return o.b;
}

__device__ __forceinline__ v8bf cvt8(const float4 f0, const float4 f1) {
  v8bf o;
  o[0] = f2bf(f0.x); o[1] = f2bf(f0.y); o[2] = f2bf(f0.z); o[3] = f2bf(f0.w);
  o[4] = f2bf(f1.x); o[5] = f2bf(f1.y); o[6] = f2bf(f1.z); o[7] = f2bf(f1.w);
  return o;
}

// Assemble a 16-bit WMMA fragment (per-lane 16 bf16) from two contiguous
// 8-element runs at kbase and kbase+16 (ISA 16-bit A/B layout:
// element i -> k = (i&7) + (i>>3)*16, +8 for lanes 16..31 via kbase).
__device__ __forceinline__ v16bf load_fragBF(const __bf16* __restrict__ row, int kbase) {
  v8bf lo = *(const v8bf*)(row + kbase);
  v8bf hi = *(const v8bf*)(row + kbase + 16);
  v16bf r;
#pragma unroll
  for (int i = 0; i < 8; ++i) { r[i] = lo[i]; r[i + 8] = hi[i]; }
  return r;
}

// f32 source -> fragment (only used in the tiny proj_dec kernel's A path)
__device__ __forceinline__ v16bf load_fragF32(const float* __restrict__ row, int kbase) {
  const float4* p = (const float4*)(row + kbase);
  const float4* q = (const float4*)(row + kbase + 16);
  v8bf lo = cvt8(p[0], p[1]);
  v8bf hi = cvt8(q[0], q[1]);
  v16bf r;
#pragma unroll
  for (int i = 0; i < 8; ++i) { r[i] = lo[i]; r[i + 8] = hi[i]; }
  return r;
}

// ---------------------------------------------------------------------------
// Kernel 0: one-shot W1 (512x1024 f32) -> bf16 copy in workspace.
// ---------------------------------------------------------------------------
__global__ void k_cvtW1(const float* __restrict__ W1, __bf16* __restrict__ W1bf) {
  const size_t base = ((size_t)blockIdx.x * 256 + threadIdx.x) * 8;
  const float4 f0 = *(const float4*)(W1 + base);
  const float4 f1 = *(const float4*)(W1 + base + 4);
  *(v8bf*)(W1bf + base) = cvt8(f0, f1);
}

// ---------------------------------------------------------------------------
// Kernel 1: proj_dec[b,h] = sum_d dh[b,d] * W1[h,d]   (128x512x512 GEMM)
// grid = 8 (M tiles of 16), block = 256 (8 waves x 4 N-tiles each)
// ---------------------------------------------------------------------------
__global__ void k_proj_dec(const float* __restrict__ dh,
                           const __bf16* __restrict__ W1bf,
                           float* __restrict__ proj_dec) {
  const int lane  = threadIdx.x & 31;
  const int wave  = threadIdx.x >> 5;
  const int m0    = blockIdx.x * 16;
  const int nlo   = lane & 15;
  const int khalf = ((lane >> 4) & 1) * 8;

  const float* arow = dh + (size_t)(m0 + nlo) * D_;

  v8f acc[4] = {};
  for (int k0 = 0; k0 < D_; k0 += 32) {
    v16bf a = load_fragF32(arow, k0 + khalf);
#pragma unroll
    for (int nt = 0; nt < 4; ++nt) {
      const int h = wave * 64 + nt * 16 + nlo;
      v16bf bfr = load_fragBF(W1bf + (size_t)h * (2 * D_), k0 + khalf); // W1_dec
      acc[nt] = __builtin_amdgcn_wmma_f32_16x16x32_bf16(
          false, a, false, bfr, (short)0, acc[nt], false, false);
    }
  }
  const int mbase = m0 + ((lane >> 4) << 3);
#pragma unroll
  for (int nt = 0; nt < 4; ++nt) {
    const int n = wave * 64 + nt * 16 + nlo;
#pragma unroll
    for (int v = 0; v < 8; ++v)
      proj_dec[(size_t)(mbase + v) * D_ + n] = acc[nt][v];
  }
}

// ---------------------------------------------------------------------------
// Kernel 2: fused scores[b,s] = W2 . tanh(enc[b,s,:]@W1_enc^T + proj_dec[b]+b1)
// grid = (S/32, B), block = 256.  A tile staged once per block into LDS (bf16),
// B streamed from pre-converted bf16 W1 (lives in L2). 8 WMMA / wave / K-step.
// ---------------------------------------------------------------------------
__global__ void k_scores(const float* __restrict__ enc,
                         const __bf16* __restrict__ W1bf,
                         const float* __restrict__ proj_dec,
                         const float* __restrict__ b1,
                         const float* __restrict__ W2,
                         const float* __restrict__ b2v,
                         float* __restrict__ scores) {
  __shared__ __align__(16) __bf16 Atile[SROWS * LDP];
  __shared__ float s_sc[SROWS];
  const int t    = threadIdx.x;
  const int lane = t & 31;
  const int wave = t >> 5;
  const int b    = blockIdx.y;
  const int s0   = blockIdx.x * SROWS;

  if (t < SROWS) s_sc[t] = 0.f;

  // --- Stage A: 32 rows x 512 cols f32 -> bf16 LDS, converted ONCE per block.
  {
    const int r  = t >> 3;
    const int cb = (t & 7) * 8;
    const float* src = enc + ((size_t)b * S_ + s0 + r) * D_;
#pragma unroll
    for (int j = 0; j < 8; ++j) {
      const int c = cb + j * 64;
      const float4 f0 = *(const float4*)(src + c);
      const float4 f1 = *(const float4*)(src + c + 4);
      *(v8bf*)(&Atile[r * LDP + c]) = cvt8(f0, f1);
    }
  }
  __syncthreads();

  const int nlo   = lane & 15;
  const int khalf = ((lane >> 4) & 1) * 8;

  v8f acc[2][4] = {};
  for (int k0 = 0; k0 < D_; k0 += 32) {
    const int kb = k0 + khalf;
    v16bf a0 = load_fragBF(&Atile[nlo * LDP], kb);          // rows s0..s0+15
    v16bf a1 = load_fragBF(&Atile[(16 + nlo) * LDP], kb);   // rows s0+16..s0+31
#pragma unroll
    for (int nt = 0; nt < 4; ++nt) {
      const int h = wave * 64 + nt * 16 + nlo;
      v16bf bfr = load_fragBF(W1bf + (size_t)h * (2 * D_) + D_, kb); // W1_enc
      acc[0][nt] = __builtin_amdgcn_wmma_f32_16x16x32_bf16(
          false, a0, false, bfr, (short)0, acc[0][nt], false, false);
      acc[1][nt] = __builtin_amdgcn_wmma_f32_16x16x32_bf16(
          false, a1, false, bfr, (short)0, acc[1][nt], false, false);
    }
  }

  // --- Epilogue: tanh + dot(W2), reduce per s-row via LDS f32 atomics.
  const float* pd  = proj_dec + (size_t)b * D_;
  const int mloc   = (lane >> 4) << 3;
#pragma unroll
  for (int mt = 0; mt < 2; ++mt) {
    float psum[8] = {0.f, 0.f, 0.f, 0.f, 0.f, 0.f, 0.f, 0.f};
#pragma unroll
    for (int nt = 0; nt < 4; ++nt) {
      const int h   = wave * 64 + nt * 16 + nlo;
      const float c = pd[h] + b1[h];
      const float w = W2[h];
#pragma unroll
      for (int v = 0; v < 8; ++v) {
        float x = acc[mt][nt][v] + c;
        x = fminf(fmaxf(x, -15.f), 15.f);
        float e = __expf(2.f * x);                               // v_exp_f32
        psum[v] += (e - 1.f) * __builtin_amdgcn_rcpf(e + 1.f) * w;
      }
    }
#pragma unroll
    for (int v = 0; v < 8; ++v)
      atomicAdd(&s_sc[mt * 16 + mloc + v], psum[v]);             // ds_add_f32
  }
  __syncthreads();
  if (t < SROWS)
    scores[(size_t)b * S_ + s0 + t] = s_sc[t] + b2v[0];
}

// ---------------------------------------------------------------------------
// Kernel 3: alpha[b,:] = softmax(scores[b,:]) ; zero-init out[b,:]
// ---------------------------------------------------------------------------
__global__ void k_softmax(const float* __restrict__ scores,
                          float* __restrict__ alpha,
                          float* __restrict__ out) {
  __shared__ float red[8];
  const int b = blockIdx.x;
  const int t = threadIdx.x;

  for (int d = t; d < D_; d += 256) out[(size_t)b * D_ + d] = 0.f;

  float v[4];
  float mx = -1e30f;
#pragma unroll
  for (int i = 0; i < 4; ++i) {
    v[i] = scores[(size_t)b * S_ + t + i * 256];
    mx = fmaxf(mx, v[i]);
  }
#pragma unroll
  for (int off = 16; off > 0; off >>= 1) mx = fmaxf(mx, __shfl_xor(mx, off, 32));
  if ((t & 31) == 0) red[t >> 5] = mx;
  __syncthreads();
  if (t == 0) {
    float m = red[0];
#pragma unroll
    for (int i = 1; i < 8; ++i) m = fmaxf(m, red[i]);
    red[0] = m;
  }
  __syncthreads();
  const float bm = red[0];
  __syncthreads();

  float sum = 0.f;
#pragma unroll
  for (int i = 0; i < 4; ++i) { v[i] = __expf(v[i] - bm); sum += v[i]; }
#pragma unroll
  for (int off = 16; off > 0; off >>= 1) sum += __shfl_xor(sum, off, 32);
  if ((t & 31) == 0) red[t >> 5] = sum;
  __syncthreads();
  if (t == 0) {
    float s = 0.f;
#pragma unroll
    for (int i = 0; i < 8; ++i) s += red[i];
    red[0] = s;
  }
  __syncthreads();
  const float inv = __builtin_amdgcn_rcpf(red[0]);
#pragma unroll
  for (int i = 0; i < 4; ++i)
    alpha[(size_t)b * S_ + t + i * 256] = v[i] * inv;
}

// ---------------------------------------------------------------------------
// Kernel 4: context[b,d] += sum_s alpha[b,s] * enc[b,s,d]  (memory-bound GEMV)
// float2 (b64) coalesced reads + global_prefetch_b8; alpha tile staged to LDS
// via the CDNA5 async global->LDS path (GLOBAL_LOAD_ASYNC_TO_LDS_B32).
// grid = (S/128, B), block = 256 (each thread owns 2 adjacent d-columns).
// ---------------------------------------------------------------------------
__global__ void k_context(const float* __restrict__ enc,
                          const float* __restrict__ alpha,
                          float* __restrict__ out) {
  __shared__ float a_s[128];
  const int b    = blockIdx.y;
  const int sBeg = blockIdx.x * 128;
  const int t    = threadIdx.x;

#if __has_builtin(__builtin_amdgcn_global_load_async_to_lds_b32)
  if (t < 128) {
    __builtin_amdgcn_global_load_async_to_lds_b32(
        (__attribute__((address_space(1))) int*)(alpha + (size_t)b * S_ + sBeg + t),
        (__attribute__((address_space(3))) int*)&a_s[t],
        0, 0);                                   // GLOBAL_LOAD_ASYNC_TO_LDS_B32
  }
#if __has_builtin(__builtin_amdgcn_s_wait_asynccnt)
  __builtin_amdgcn_s_wait_asynccnt(0);
#else
  asm volatile("s_wait_asynccnt 0x0" ::: "memory");
#endif
#else
  if (t < 128)
    a_s[t] = alpha[(size_t)b * S_ + sBeg + t];
#endif
  __syncthreads();

  const float2* ep = (const float2*)(enc + ((size_t)b * S_ + sBeg) * D_) + t;
  float2 acc; acc.x = 0.f; acc.y = 0.f;
  for (int s = 0; s < 128; s += 4) {
    __builtin_prefetch((const float*)(ep + (size_t)(s + 8) * (D_ / 2)), 0, 1);
    float2 e0 = ep[(size_t)(s + 0) * (D_ / 2)];
    float2 e1 = ep[(size_t)(s + 1) * (D_ / 2)];
    float2 e2 = ep[(size_t)(s + 2) * (D_ / 2)];
    float2 e3 = ep[(size_t)(s + 3) * (D_ / 2)];
    acc.x = fmaf(a_s[s + 0], e0.x, acc.x); acc.y = fmaf(a_s[s + 0], e0.y, acc.y);
    acc.x = fmaf(a_s[s + 1], e1.x, acc.x); acc.y = fmaf(a_s[s + 1], e1.y, acc.y);
    acc.x = fmaf(a_s[s + 2], e2.x, acc.x); acc.y = fmaf(a_s[s + 2], e2.y, acc.y);
    acc.x = fmaf(a_s[s + 3], e3.x, acc.x); acc.y = fmaf(a_s[s + 3], e3.y, acc.y);
  }
  atomicAdd(&out[(size_t)b * D_ + 2 * t + 0], acc.x);   // global_atomic_add_f32
  atomicAdd(&out[(size_t)b * D_ + 2 * t + 1], acc.y);
}

// ---------------------------------------------------------------------------
extern "C" void kernel_launch(void* const* d_in, const int* in_sizes, int n_in,
                              void* d_out, int out_size, void* d_ws, size_t ws_size,
                              hipStream_t stream) {
  const float* dh  = (const float*)d_in[0];   // (B, D)
  const float* enc = (const float*)d_in[1];   // (B, S, D)
  const float* W1  = (const float*)d_in[2];   // (D, 2D)
  const float* b1  = (const float*)d_in[3];   // (D,)
  const float* W2  = (const float*)d_in[4];   // (1, D)
  const float* b2  = (const float*)d_in[5];   // (1,)
  float* out = (float*)d_out;                 // (B, D)

  __bf16* W1bf    = (__bf16*)d_ws;                       // 512*1024 bf16 (1 MB)
  float* proj_dec = (float*)(W1bf + (size_t)D_ * 2 * D_);
  float* scores   = proj_dec + (size_t)B_ * D_;
  float* alpha    = scores   + (size_t)B_ * S_;

  k_cvtW1   <<<dim3(256), 256, 0, stream>>>(W1, W1bf);
  k_proj_dec<<<dim3(B_ / 16), 256, 0, stream>>>(dh, W1bf, proj_dec);
  k_scores  <<<dim3(S_ / SROWS, B_), 256, 0, stream>>>(enc, W1bf, proj_dec, b1, W2, b2, scores);
  k_softmax <<<dim3(B_), 256, 0, stream>>>(scores, alpha, out);
  k_context <<<dim3(S_ / 128, B_), 256, 0, stream>>>(enc, alpha, out);
}
